// _quantized_linear_25769803776177
// MI455X (gfx1250) — matrix operand order repaired and checked by reading, not compiled, not run on hardware
//
#include <hip/hip_runtime.h>

typedef __attribute__((ext_vector_type(16))) _Float16 v16h;
typedef __attribute__((ext_vector_type(8)))  _Float16 v8h;
typedef __attribute__((ext_vector_type(4)))  _Float16 v4h;
typedef __attribute__((ext_vector_type(2)))  __fp16   v2hp;  // cvt_pkrtz result type
typedef __attribute__((ext_vector_type(8)))  float    v8f;
typedef __attribute__((ext_vector_type(4)))  float    v4f;
typedef __attribute__((ext_vector_type(4)))  int      v4i;

#define M_DIM 2048
#define K_DIM 4096
#define N_DIM 11008
#define GSIZE 128
#define G_DIM (K_DIM / GSIZE)   // 32 groups

#define BM 128
#define BN 256
#define BK 64
#define LDK 72                  // padded LDS K-stride in halves (144 B)

union FragU { v16h v; v8h h[2]; };
union H4   { v4h  v; v2hp h[2]; };

__global__ __launch_bounds__(256)
void _quantized_linear_wmma_kernel(const float* __restrict__ x,
                                   const int*   __restrict__ wq,
                                   const float* __restrict__ ws,
                                   const int*   __restrict__ wz,
                                   float* __restrict__ out)
{
    __shared__ _Float16 Alds[BM][LDK];   // x tile, f16, row-major [m][k]
    __shared__ _Float16 Blds[BN][LDK];   // dequantized weight tile [n][k]

    const int tid  = threadIdx.x;
    const int lane = tid & 31;
    const int wave = tid >> 5;        // 0..7
    const int wm   = wave >> 2;       // 0..1 : 64-row band in M
    const int wn   = wave & 3;        // 0..3 : 64-col band in N

    const int bn0 = blockIdx.x * BN;
    const int bm0 = blockIdx.y * BM;

    // loader decomposition: 16 rows x 16 col-groups (4 elems) per pass
    const int lr = tid >> 4;          // 0..15
    const int lc = tid & 15;          // 0..15 -> 4-element column group

    v8f acc[4][4];
    {
        v8f zero = {};
        #pragma unroll
        for (int i = 0; i < 4; ++i)
            #pragma unroll
            for (int j = 0; j < 4; ++j)
                acc[i][j] = zero;
    }

    for (int kb = 0; kb < K_DIM; kb += BK) {
        const int g = kb >> 7;   // group index (64-chunk stays inside one 128-group)

        __syncthreads();         // previous compute done with LDS

        if (kb + BK < K_DIM) {   // speculative prefetch of next K-chunk
            __builtin_prefetch(x  + (size_t)(bm0 + lr) * K_DIM + (kb + BK) + lc * 4, 0, 1);
            __builtin_prefetch(wq + (size_t)(bn0 + lr) * K_DIM + (kb + BK) + lc * 4, 0, 1);
        }

        // ---- x tile: 128 rows, f32x4 -> pk f16, stage to LDS ----
        #pragma unroll
        for (int p = 0; p < 8; ++p) {
            const int row = lr + p * 16;
            const float* src = x + (size_t)(bm0 + row) * K_DIM + kb + lc * 4;
            v4f v = *(const v4f*)src;
            H4 h;
            h.h[0] = __builtin_amdgcn_cvt_pkrtz(v.x, v.y);
            h.h[1] = __builtin_amdgcn_cvt_pkrtz(v.z, v.w);
            *(v4h*)&Alds[row][lc * 4] = h.v;
        }

        // ---- weight tile: 256 rows, int32x4 -> fma(q,s,-z*s) -> pk f16 ----
        #pragma unroll
        for (int p = 0; p < 16; ++p) {
            const int row = lr + p * 16;
            const int n = bn0 + row;
            const int* srcq = wq + (size_t)n * K_DIM + kb + lc * 4;
            v4i q = *(const v4i*)srcq;
            const float s  = ws[n * G_DIM + g];
            const float zs = -(float)wz[n * G_DIM + g] * s;
            const float f0 = __builtin_fmaf((float)q.x, s, zs);
            const float f1 = __builtin_fmaf((float)q.y, s, zs);
            const float f2 = __builtin_fmaf((float)q.z, s, zs);
            const float f3 = __builtin_fmaf((float)q.w, s, zs);
            H4 h;
            h.h[0] = __builtin_amdgcn_cvt_pkrtz(f0, f1);
            h.h[1] = __builtin_amdgcn_cvt_pkrtz(f2, f3);
            *(v4h*)&Blds[row][lc * 4] = h.v;
        }

        __syncthreads();         // tiles visible to all waves

        #pragma unroll
        for (int ks = 0; ks < 2; ++ks) {
            const int k0 = ks * 32;

            // A fragments: 16x32 f16; lane = M row; lanes 0-15 hold K [0..8)+[16..24),
            // lanes 16-31 hold K [8..16)+[24..32)
            FragU a[4];
            const int arow = wm * 64 + (lane & 15);
            const int ac0  = k0 + ((lane >> 4) << 3);
            #pragma unroll
            for (int mi = 0; mi < 4; ++mi) {
                const int r = arow + mi * 16;
                a[mi].h[0] = *(const v8h*)&Alds[r][ac0];
                a[mi].h[1] = *(const v8h*)&Alds[r][ac0 + 16];
            }

            // B fragments: 32x16 f16; lane = N column; lanes 0-15 hold K [0..8)+[16..24),
            // lanes 16-31 hold K [8..16)+[24..32)
            FragU b[4];
            const int brow = wn * 64 + (lane & 15);
            const int bc0  = k0 + ((lane >> 4) << 3);
            #pragma unroll
            for (int ni = 0; ni < 4; ++ni) {
                const int r = brow + ni * 16;
                b[ni].h[0] = *(const v8h*)&Blds[r][bc0];
                b[ni].h[1] = *(const v8h*)&Blds[r][bc0 + 16];
            }

            #pragma unroll
            for (int mi = 0; mi < 4; ++mi)
                #pragma unroll
                for (int ni = 0; ni < 4; ++ni)
                    acc[mi][ni] = __builtin_amdgcn_wmma_f32_16x16x32_f16(
                        false, a[mi].v, false, b[ni].v,
                        (short)0, acc[mi][ni], false, false);
        }
    }

    // ---- epilogue: C/D layout: lane%16 = N col, M row = vgpr + (lane/16)*8 ----
    const int col0 = bn0 + wn * 64 + (lane & 15);
    const int row0 = bm0 + wm * 64 + ((lane >> 4) << 3);
    #pragma unroll
    for (int mi = 0; mi < 4; ++mi) {
        #pragma unroll
        for (int ni = 0; ni < 4; ++ni) {
            float* dst = out + (size_t)(row0 + mi * 16) * N_DIM + (col0 + ni * 16);
            v8f c = acc[mi][ni];
            #pragma unroll
            for (int i = 0; i < 8; ++i)
                dst[(size_t)i * N_DIM] = c[i];
        }
    }
}

extern "C" void kernel_launch(void* const* d_in, const int* in_sizes, int n_in,
                              void* d_out, int out_size, void* d_ws, size_t ws_size,
                              hipStream_t stream) {
    const float* x  = (const float*)d_in[0];
    const int*   wq = (const int*)d_in[1];
    const float* ws = (const float*)d_in[2];
    const int*   wz = (const int*)d_in[3];
    // d_in[4] = group_size scalar (=128), baked in as a compile-time constant
    float* out = (float*)d_out;

    dim3 grid(N_DIM / BN, M_DIM / BM);   // 43 x 16 blocks
    _quantized_linear_wmma_kernel<<<grid, dim3(256), 0, stream>>>(x, wq, ws, wz, out);
}
